// Interaction_17600775979354
// MI455X (gfx1250) — compile-verified, matrix operand order checked by reading, not run on hardware
//
#include <hip/hip_runtime.h>
#include <hip/hip_bf16.h>

#define C_CH   64
#define R_RB   8
#define H_HID  64
#define NPATH  11
#define NSLOT  13                 // 1 + 3 + 9 irrep slots
#define WCOLS  (NPATH * C_CH)     // 704
#define UROW   (NSLOT * C_CH)     // 832 floats per node row

typedef __attribute__((ext_vector_type(16))) __bf16 bf16x16;
typedef __attribute__((ext_vector_type(8)))  float  f32x8;

// Packed bf16 B-fragment tiles (each tile = 32 lanes x 16 bf16 = 1 KB):
//   Wr0: 4 tiles   [0,4)    (ntile 0..3, K padded 8->32)
//   Wr1: 8 tiles   [4,12)   (index = ntile*2 + ktile)
//   Wr2: 8 tiles   [12,20)
//   Wr3: 88 tiles  [20,108) (ntile 0..43)
#define TILE_WR0 0
#define TILE_WR1 4
#define TILE_WR2 12
#define TILE_WR3 20
#define NTILES   108

// ---------------------------------------------------------------------------
// WMMA fragment helpers (CDNA5 16x16x32 bf16 VGPR layouts, wave32)
// ---------------------------------------------------------------------------
__device__ inline bf16x16 load_a_frag(const __bf16* act, int ldk, int lane, int kbase) {
  // A 16x32: lane L -> row m = L&15; half = L>>4 selects K groups {0..7|8..15}
  // reg r holds K pair: k = (r>>2)*16 + half*8 + (r&3)*2
  const int m = lane & 15, half = lane >> 4;
  const __bf16* row = act + m * ldk + kbase + half * 8;
  bf16x16 f;
#pragma unroll
  for (int r = 0; r < 8; ++r) {
    const int k = (r >> 2) * 16 + (r & 3) * 2;
    f[2 * r]     = row[k];
    f[2 * r + 1] = row[k + 1];
  }
  return f;
}

__device__ inline bf16x16 load_b_frag(const unsigned short* wpack, int tile, int lane) {
  // pre-swizzled: one contiguous 32B vector per lane
  return *(const bf16x16*)(wpack + ((size_t)tile * 32 + lane) * 16);
}

// Fast SiLU: x * sigmoid(x) with hardware v_rcp_f32 + v_exp_f32
__device__ inline float silu_f(float x) {
  return x * __builtin_amdgcn_rcpf(1.0f + __expf(-x));
}

__device__ inline void atomic_add_f32(float* p, float v) {
  (void)__hip_atomic_fetch_add(p, v, __ATOMIC_RELAXED, __HIP_MEMORY_SCOPE_AGENT);
}

// ---------------------------------------------------------------------------
// Kernel 0: pack radial-MLP weights f32 -> bf16 in WMMA-B fragment order
// ---------------------------------------------------------------------------
__global__ void pack_weights_kernel(const float* __restrict__ Wr0,
                                    const float* __restrict__ Wr1,
                                    const float* __restrict__ Wr2,
                                    const float* __restrict__ Wr3,
                                    unsigned short* __restrict__ wpack) {
  const int tile = blockIdx.x;
  const float* W; int Krows, ncol, ntile, ktile;
  if (tile < TILE_WR1)      { W = Wr0; Krows = R_RB;  ncol = H_HID; ntile = tile; ktile = 0; }
  else if (tile < TILE_WR2) { int i = tile - TILE_WR1; W = Wr1; Krows = H_HID; ncol = H_HID; ntile = i >> 1; ktile = i & 1; }
  else if (tile < TILE_WR3) { int i = tile - TILE_WR2; W = Wr2; Krows = H_HID; ncol = H_HID; ntile = i >> 1; ktile = i & 1; }
  else                      { int i = tile - TILE_WR3; W = Wr3; Krows = H_HID; ncol = WCOLS; ntile = i >> 1; ktile = i & 1; }
  for (int idx = threadIdx.x; idx < 512; idx += blockDim.x) {
    const int lane = idx >> 4, v = idx & 15;
    const int n = lane & 15, half = lane >> 4;
    const int r = v >> 1;
    const int k = ktile * 32 + (r >> 2) * 16 + half * 8 + (r & 3) * 2 + (v & 1);
    const float val = (k < Krows) ? W[k * ncol + ntile * 16 + n] : 0.0f;
    wpack[((size_t)tile * 32 + lane) * 16 + v] =
        __builtin_bit_cast(unsigned short, (__bf16)val);
  }
}

// ---------------------------------------------------------------------------
// Kernel 1: linear_up — per-irrep channel mix, u[n][j][c] (memory-bound VALU)
// ---------------------------------------------------------------------------
__global__ __launch_bounds__(256)
void linear_up_kernel(const float* __restrict__ t0, const float* __restrict__ t1,
                      const float* __restrict__ t2,
                      const float* __restrict__ W0, const float* __restrict__ W1,
                      const float* __restrict__ W2,
                      float* __restrict__ u) {
  __shared__ float tin[NSLOT][C_CH];
  const int n = blockIdx.x;
  for (int idx = threadIdx.x; idx < NSLOT * C_CH; idx += 256) {
    const int j = idx / C_CH, k = idx % C_CH;
    float v;
    if (j == 0)     v = t0[n * C_CH + k];
    else if (j < 4) v = t1[(size_t)(n * C_CH + k) * 3 + (j - 1)];
    else            v = t2[(size_t)(n * C_CH + k) * 9 + (j - 4)];
    tin[j][k] = v;
  }
  __syncthreads();
  for (int idx = threadIdx.x; idx < NSLOT * C_CH; idx += 256) {
    const int j = idx / C_CH, c = idx % C_CH;
    const float* W = (j == 0) ? W0 : ((j < 4) ? W1 : W2);
    float acc = 0.f;
#pragma unroll 8
    for (int k = 0; k < C_CH; ++k) acc += tin[j][k] * W[k * C_CH + c];
    u[(size_t)n * UROW + j * C_CH + c] = acc;
  }
}

// ---------------------------------------------------------------------------
// Kernel 2: fused edge kernel — radial MLP (WMMA bf16) + tensor product + scatter
// 16 edges per workgroup, 8 wave32s. w[16][704] lives entirely in LDS.
// ---------------------------------------------------------------------------
__global__ __launch_bounds__(256)
void edge_kernel(const float* __restrict__ u,
                 const float* __restrict__ a0, const float* __restrict__ a1,
                 const float* __restrict__ a2,
                 const float* __restrict__ ef, const float* __restrict__ cut,
                 const unsigned short* __restrict__ wpack,
                 const int* __restrict__ eidx,
                 float* __restrict__ macc, int nE) {
  __shared__ __bf16 actA[16][32];        // layer-1 input, K padded to 32
  __shared__ __bf16 actB[16][H_HID];     // ping
  __shared__ __bf16 actC[16][H_HID];     // pong
  __shared__ float  wbuf[16][WCOLS];     // path weights, cutoff-gated (45 KB)
  __shared__ float  aLDS[16][NSLOT];
  __shared__ float  cutLDS[16];
  __shared__ int    srcLDS[16], dstLDS[16];

  const int tid  = threadIdx.x;
  const int lane = tid & 31;
  const int wv   = tid >> 5;
  const int eb   = blockIdx.x * 16;

  // ---- phase 0: stage edge metadata into LDS ----
  if (tid < 16) {
    const int ge = eb + tid;
    int s = 0, d = 0; float cv = 0.f;
    if (ge < nE) { s = eidx[ge]; d = eidx[nE + ge]; cv = cut[ge]; }
    srcLDS[tid] = s; dstLDS[tid] = d; cutLDS[tid] = cv;
  }
  for (int idx = tid; idx < 16 * NSLOT; idx += 256) {
    const int e = idx / NSLOT, j = idx % NSLOT;
    const int ge = eb + e;
    float v = 0.f;
    if (ge < nE) {
      if (j == 0)     v = a0[ge];
      else if (j < 4) v = a1[(size_t)ge * 3 + (j - 1)];
      else            v = a2[(size_t)ge * 9 + (j - 4)];
    }
    aLDS[e][j] = v;
  }
  for (int idx = tid; idx < 16 * 32; idx += 256) {
    const int e = idx >> 5, k = idx & 31;
    const int ge = eb + e;
    const float v = (k < R_RB && ge < nE) ? ef[(size_t)ge * R_RB + k] : 0.f;
    actA[e][k] = (__bf16)v;
  }
  __syncthreads();

  // warm L2 for the gathered sender rows while the MLP runs
  for (int idx = tid; idx < 16 * 16; idx += 256) {
    const int e = idx >> 4, seg = idx & 15;
    const char* p = (const char*)(u + (size_t)srcLDS[e] * UROW) + seg * 128;
    __builtin_prefetch(p, 0, 1);
  }

  // ---- phase 1: radial MLP, f32 accumulate via v_wmma_f32_16x16x32_bf16 ----
  // layer 1: [16 x 8(pad32)] x Wr0 -> actB
  if (wv < 4) {
    f32x8 acc = {};
    bf16x16 afr = load_a_frag(&actA[0][0], 32, lane, 0);
    bf16x16 bfr = load_b_frag(wpack, TILE_WR0 + wv, lane);
    acc = __builtin_amdgcn_wmma_f32_16x16x32_bf16(false, afr, false, bfr,
                                                  (short)0, acc, false, false);
    const int n = lane & 15, half = lane >> 4;
#pragma unroll
    for (int r = 0; r < 8; ++r)
      actB[r + 8 * half][wv * 16 + n] = (__bf16)silu_f(acc[r]);
  }
  __syncthreads();
  // layer 2: actB x Wr1 -> actC
  if (wv < 4) {
    f32x8 acc = {};
#pragma unroll
    for (int kt = 0; kt < 2; ++kt) {
      bf16x16 afr = load_a_frag(&actB[0][0], H_HID, lane, kt * 32);
      bf16x16 bfr = load_b_frag(wpack, TILE_WR1 + wv * 2 + kt, lane);
      acc = __builtin_amdgcn_wmma_f32_16x16x32_bf16(false, afr, false, bfr,
                                                    (short)0, acc, false, false);
    }
    const int n = lane & 15, half = lane >> 4;
#pragma unroll
    for (int r = 0; r < 8; ++r)
      actC[r + 8 * half][wv * 16 + n] = (__bf16)silu_f(acc[r]);
  }
  __syncthreads();
  // layer 3: actC x Wr2 -> actB
  if (wv < 4) {
    f32x8 acc = {};
#pragma unroll
    for (int kt = 0; kt < 2; ++kt) {
      bf16x16 afr = load_a_frag(&actC[0][0], H_HID, lane, kt * 32);
      bf16x16 bfr = load_b_frag(wpack, TILE_WR2 + wv * 2 + kt, lane);
      acc = __builtin_amdgcn_wmma_f32_16x16x32_bf16(false, afr, false, bfr,
                                                    (short)0, acc, false, false);
    }
    const int n = lane & 15, half = lane >> 4;
#pragma unroll
    for (int r = 0; r < 8; ++r)
      actB[r + 8 * half][wv * 16 + n] = (__bf16)silu_f(acc[r]);
  }
  __syncthreads();
  // layer 4: actB x Wr3 -> wbuf, gated by cutoff. 44 N-tiles over 8 waves.
  // A fragments are loop-invariant: load once, keep in VGPRs.
  {
    const bf16x16 afr0 = load_a_frag(&actB[0][0], H_HID, lane, 0);
    const bf16x16 afr1 = load_a_frag(&actB[0][0], H_HID, lane, 32);
    const int n = lane & 15, half = lane >> 4;
    for (int t = wv; t < 44; t += 8) {
      f32x8 acc = {};
      bf16x16 b0 = load_b_frag(wpack, TILE_WR3 + t * 2 + 0, lane);
      bf16x16 b1 = load_b_frag(wpack, TILE_WR3 + t * 2 + 1, lane);
      acc = __builtin_amdgcn_wmma_f32_16x16x32_bf16(false, afr0, false, b0,
                                                    (short)0, acc, false, false);
      acc = __builtin_amdgcn_wmma_f32_16x16x32_bf16(false, afr1, false, b1,
                                                    (short)0, acc, false, false);
#pragma unroll
      for (int r = 0; r < 8; ++r) {
        const int m = r + 8 * half;
        wbuf[m][t * 16 + n] = acc[r] * cutLDS[m];
      }
    }
  }
  __syncthreads();

  // ---- phase 2: Cartesian tensor-product paths + scatter-add ----
  for (int it = 0; it < 4; ++it) {
    const int idx = it * 256 + tid;
    const int e = idx >> 6, c = idx & 63;
    const int ge = eb + e;
    if (ge >= nE) continue;
    const int src = srcLDS[e], dst = dstLDS[e];
    const float* up = u + (size_t)src * UROW + c;
    const float s0 = up[0];
    float s1[3], s2[9];
#pragma unroll
    for (int i = 0; i < 3; ++i) s1[i] = up[(1 + i) * C_CH];
#pragma unroll
    for (int i = 0; i < 9; ++i) s2[i] = up[(4 + i) * C_CH];
    const float A0 = aLDS[e][0];
    float A1[3], A2[9];
#pragma unroll
    for (int i = 0; i < 3; ++i) A1[i] = aLDS[e][1 + i];
#pragma unroll
    for (int i = 0; i < 9; ++i) A2[i] = aLDS[e][4 + i];
    const float* wrow = &wbuf[e][0];
    float wq[NPATH];
#pragma unroll
    for (int p = 0; p < NPATH; ++p) wq[p] = wrow[p * C_CH + c];
    float* mp = macc + (size_t)dst * UROW + c;

    // lout = 0
    const float dot11 = A1[0] * s1[0] + A1[1] * s1[1] + A1[2] * s1[2];
    float dot22 = 0.f;
#pragma unroll
    for (int i = 0; i < 9; ++i) dot22 += A2[i] * s2[i];
    atomic_add_f32(mp, wq[0] * (A0 * s0) + wq[1] * dot11 + wq[2] * dot22);

    // lout = 1
#pragma unroll
    for (int i = 0; i < 3; ++i) {
      const float t5 = A2[i*3+0]*s1[0] + A2[i*3+1]*s1[1] + A2[i*3+2]*s1[2];
      const float t6 = A1[0]*s2[i*3+0] + A1[1]*s2[i*3+1] + A1[2]*s2[i*3+2];
      atomic_add_f32(mp + (1 + i) * C_CH,
                     wq[3]*(A1[i]*s0) + wq[4]*(A0*s1[i]) + wq[5]*t5 + wq[6]*t6);
    }

    // lout = 2
#pragma unroll
    for (int i = 0; i < 3; ++i)
#pragma unroll
      for (int j = 0; j < 3; ++j) {
        const float t10 = A2[i*3+0]*s2[0+j] + A2[i*3+1]*s2[3+j] + A2[i*3+2]*s2[6+j];
        atomic_add_f32(mp + (4 + i*3 + j) * C_CH,
                       wq[7]*(A1[i]*s1[j]) + wq[8]*(A2[i*3+j]*s0) +
                       wq[9]*(A0*s2[i*3+j]) + wq[10]*t10);
      }
  }
}

// ---------------------------------------------------------------------------
// Kernel 3: 1/avg_neighbors scale + D_{l,l,1} symmetrization epilogue
// ---------------------------------------------------------------------------
__global__ __launch_bounds__(256)
void finalize_kernel(const float* __restrict__ macc,
                     const float* __restrict__ D0, const float* __restrict__ D1,
                     const float* __restrict__ D2,
                     float* __restrict__ out, int nN) {
  const int idx = blockIdx.x * blockDim.x + threadIdx.x;
  if (idx >= nN * C_CH) return;
  const float inv = 1.0f / 32.0f;
  const int n = idx >> 6, c = idx & 63;
  const float* mb = macc + (size_t)n * UROW + c;
  const float m0 = mb[0] * inv;
  float m1[3], m2[9];
#pragma unroll
  for (int i = 0; i < 3; ++i) m1[i] = mb[(1 + i) * C_CH] * inv;
#pragma unroll
  for (int i = 0; i < 9; ++i) m2[i] = mb[(4 + i) * C_CH] * inv;

  out[idx] = m0 * D0[0];
  float* o1 = out + (size_t)nN * C_CH + (size_t)idx * 3;
#pragma unroll
  for (int j = 0; j < 3; ++j) {
    float s = 0.f;
#pragma unroll
    for (int i = 0; i < 3; ++i) s += m1[i] * D1[i * 3 + j];
    o1[j] = s;
  }
  float* o2 = out + (size_t)nN * C_CH * 4 + (size_t)idx * 9;
#pragma unroll
  for (int j = 0; j < 9; ++j) {
    float s = 0.f;
#pragma unroll
    for (int i = 0; i < 9; ++i) s += m2[i] * D2[i * 9 + j];
    o2[j] = s;
  }
}

// ---------------------------------------------------------------------------
extern "C" void kernel_launch(void* const* d_in, const int* in_sizes, int n_in,
                              void* d_out, int out_size, void* d_ws, size_t ws_size,
                              hipStream_t stream) {
  const float* t0  = (const float*)d_in[0];
  const float* t1  = (const float*)d_in[1];
  const float* t2  = (const float*)d_in[2];
  const float* a0  = (const float*)d_in[3];
  const float* a1  = (const float*)d_in[4];
  const float* a2  = (const float*)d_in[5];
  const float* ef  = (const float*)d_in[6];
  const float* cut = (const float*)d_in[7];
  const float* W0  = (const float*)d_in[8];
  const float* W1  = (const float*)d_in[9];
  const float* W2  = (const float*)d_in[10];
  const float* Wr0 = (const float*)d_in[11];
  const float* Wr1 = (const float*)d_in[12];
  const float* Wr2 = (const float*)d_in[13];
  const float* Wr3 = (const float*)d_in[14];
  const float* D0  = (const float*)d_in[15];
  const float* D1  = (const float*)d_in[16];
  const float* D2  = (const float*)d_in[17];
  const int* eidx  = (const int*)d_in[18];

  const int nE = in_sizes[3];          // a0 has E elements
  const int nN = in_sizes[0] / C_CH;   // t0 has N*C elements
  float* out = (float*)d_out;

  char* ws = (char*)d_ws;
  const size_t uBytes = (size_t)nN * UROW * sizeof(float);
  float* u    = (float*)ws;
  float* macc = (float*)(ws + uBytes);
  unsigned short* wpack = (unsigned short*)(ws + 2 * uBytes);

  pack_weights_kernel<<<NTILES, 256, 0, stream>>>(Wr0, Wr1, Wr2, Wr3, wpack);
  linear_up_kernel<<<nN, 256, 0, stream>>>(t0, t1, t2, W0, W1, W2, u);
  (void)hipMemsetAsync(macc, 0, uBytes, stream);
  edge_kernel<<<(nE + 15) / 16, 256, 0, stream>>>(u, a0, a1, a2, ef, cut,
                                                  wpack, eidx, macc, nE);
  finalize_kernel<<<(nN * C_CH + 255) / 256, 256, 0, stream>>>(macc, D0, D1, D2,
                                                               out, nN);
}
